// LSTM_Encoder_43422119362824
// MI455X (gfx1250) — matrix operand order repaired and checked by reading, not compiled, not run on hardware
//
#include <hip/hip_runtime.h>
#include <hip/hip_bf16.h>
#include <stdint.h>

// Problem constants (match reference)
#define SEQ    4096
#define EMBED  512
#define HIDDEN 1024
#define G4H    (4 * HIDDEN)   // 4096 gate rows

typedef __attribute__((ext_vector_type(16))) __bf16 v16bf;
typedef __attribute__((ext_vector_type(8)))  float  v8f;

// ---- workspace layout (bytes) ----
#define OFF_XG   0ull
#define OFF_X    (OFF_XG   + 4ull * SEQ * G4H)
#define OFF_WIH  (OFF_X    + 2ull * SEQ * EMBED)
#define OFF_WHH  (OFF_WIH  + 2ull * (unsigned long long)G4H * EMBED)
#define OFF_BIAS (OFF_WHH  + 2ull * (unsigned long long)G4H * HIDDEN)
#define OFF_H    (OFF_BIAS + 4ull * G4H)
#define OFF_BAR  (OFF_H    + 4ull * 2 * HIDDEN)

__device__ __forceinline__ unsigned short f2bf(float f) {
  unsigned u = __builtin_bit_cast(unsigned, f);
  unsigned r = u + 0x7FFFu + ((u >> 16) & 1u);   // round-to-nearest-even
  return (unsigned short)(r >> 16);
}

__device__ __forceinline__ float bf_lo(unsigned u) {
  return __builtin_bit_cast(float, u << 16);
}
__device__ __forceinline__ float bf_hi(unsigned u) {
  return __builtin_bit_cast(float, u & 0xFFFF0000u);
}

// ---------------------------------------------------------------------------
// Kernel 1: prep — bias sum, bf16 conversions, embedding gather, zero h & bar
// ---------------------------------------------------------------------------
__global__ void lstm_prep(const int* __restrict__ tok, const float* __restrict__ emb,
                          const float* __restrict__ Wih, const float* __restrict__ Whh,
                          const float* __restrict__ bih, const float* __restrict__ bhh,
                          unsigned short* __restrict__ xbf,
                          unsigned short* __restrict__ wihbf,
                          unsigned short* __restrict__ whhbf,
                          float* __restrict__ bias, float* __restrict__ hbuf,
                          unsigned* __restrict__ bar) {
  const long NB  = G4H;
  const long NW1 = (long)G4H * EMBED;
  const long NW2 = (long)G4H * HIDDEN;
  const long NX  = (long)SEQ * EMBED;
  const long NH  = 2 * HIDDEN;
  const long TOTAL = NB + NW1 + NW2 + NX + NH + 2;
  for (long i = blockIdx.x * (long)blockDim.x + threadIdx.x; i < TOTAL;
       i += (long)gridDim.x * blockDim.x) {
    long r = i;
    if (r < NB)  { bias[r] = bih[r] + bhh[r];            continue; } r -= NB;
    if (r < NW1) { wihbf[r] = f2bf(Wih[r]);              continue; } r -= NW1;
    if (r < NW2) { whhbf[r] = f2bf(Whh[r]);              continue; } r -= NW2;
    if (r < NX)  { long t = r / EMBED, e = r - t * EMBED;
                   xbf[r] = f2bf(emb[(long)tok[t] * EMBED + e]); continue; } r -= NX;
    if (r < NH)  { hbuf[r] = 0.0f;                       continue; } r -= NH;
    bar[r] = 0u;
  }
}

// ---------------------------------------------------------------------------
// Kernel 2: xg = x @ W_ih^T + (b_ih + b_hh) via V_WMMA_F32_16X16X32_BF16.
// 2x2 register blocking: each wave computes a 32x32 output tile with 4
// accumulators; per k-step loads 2 A-frags + 2 B-frags, issues 4 WMMAs
// (16 FLOP/B from L1 instead of 8). Block = 8 waves -> 32x256 tile.
// Grid: (SEQ/32, 4H/256).
// ---------------------------------------------------------------------------
__global__ void lstm_gemm_xg(const unsigned short* __restrict__ X,
                             const unsigned short* __restrict__ W,
                             const float* __restrict__ bias,
                             float* __restrict__ xg) {
  const int tbase = blockIdx.x * 32;
  const int wave  = threadIdx.x >> 5;
  const int lane  = threadIdx.x & 31;
  const int jbase = blockIdx.y * 256 + wave * 32;
  const int n  = lane & 15;
  const int hi = lane >> 4;

  // A (16x32 bf16): lanes 0-15 rows M=0..15 K=[k..k+15];
  //                 lanes 16-31 same rows K=[k+16..k+31]
  const unsigned short* arow0 = X + (size_t)(tbase + n) * EMBED + hi * 16;
  const unsigned short* arow1 = arow0 + 16 * EMBED;
  // B (32x16 bf16): lanes 0-15 col N K=[k..k+15]; lanes 16-31 K=[k+16..k+31]
  const unsigned short* brow0 = W + (size_t)(jbase + n) * EMBED + hi * 16;
  const unsigned short* brow1 = brow0 + 16 * EMBED;

  v8f acc00, acc01, acc10, acc11;
  {
    float b0 = bias[jbase + n];
    float b1 = bias[jbase + 16 + n];
#pragma unroll
    for (int r = 0; r < 8; ++r) {
      acc00[r] = b0; acc01[r] = b1; acc10[r] = b0; acc11[r] = b1;
    }
  }

#pragma unroll 2
  for (int k = 0; k < EMBED; k += 32) {
    union { v16bf v; uint4 q[2]; } A0, A1, B0, B1;
    A0.q[0] = *(const uint4*)(arow0 + k);
    A0.q[1] = *(const uint4*)(arow0 + k + 8);
    A1.q[0] = *(const uint4*)(arow1 + k);
    A1.q[1] = *(const uint4*)(arow1 + k + 8);
    B0.q[0] = *(const uint4*)(brow0 + k);
    B0.q[1] = *(const uint4*)(brow0 + k + 8);
    B1.q[0] = *(const uint4*)(brow1 + k);
    B1.q[1] = *(const uint4*)(brow1 + k + 8);
    acc00 = __builtin_amdgcn_wmma_f32_16x16x32_bf16(false, A0.v, false, B0.v,
                                                    (short)0, acc00, false, false);
    acc01 = __builtin_amdgcn_wmma_f32_16x16x32_bf16(false, A0.v, false, B1.v,
                                                    (short)0, acc01, false, false);
    acc10 = __builtin_amdgcn_wmma_f32_16x16x32_bf16(false, A1.v, false, B0.v,
                                                    (short)0, acc10, false, false);
    acc11 = __builtin_amdgcn_wmma_f32_16x16x32_bf16(false, A1.v, false, B1.v,
                                                    (short)0, acc11, false, false);
  }

  // D layout: lane 0-15 col=lane, VGPR r -> row tbase+r; lane 16-31 -> row +8+r
  float* o0 = xg + (size_t)(tbase + hi * 8) * G4H + (jbase + n);
  float* o1 = o0 + (size_t)16 * G4H;
#pragma unroll
  for (int r = 0; r < 8; ++r) {
    o0[(size_t)r * G4H]      = acc00[r];
    o0[(size_t)r * G4H + 16] = acc01[r];
    o1[(size_t)r * G4H]      = acc10[r];
    o1[(size_t)r * G4H + 16] = acc11[r];
  }
}

// ---------------------------------------------------------------------------
// Global sense-reversing barrier (device scope)
// ---------------------------------------------------------------------------
__device__ __forceinline__ void gbar(unsigned* bar, unsigned nwg) {
  __syncthreads();
  if (threadIdx.x == 0) {
    __threadfence();
    unsigned e = __hip_atomic_load(&bar[1], __ATOMIC_RELAXED, __HIP_MEMORY_SCOPE_AGENT);
    unsigned a = __hip_atomic_fetch_add(&bar[0], 1u, __ATOMIC_ACQ_REL, __HIP_MEMORY_SCOPE_AGENT);
    if (a == nwg - 1) {
      __hip_atomic_store(&bar[0], 0u, __ATOMIC_RELAXED, __HIP_MEMORY_SCOPE_AGENT);
      __hip_atomic_fetch_add(&bar[1], 1u, __ATOMIC_RELEASE, __HIP_MEMORY_SCOPE_AGENT);
    } else {
      while (__hip_atomic_load(&bar[1], __ATOMIC_ACQUIRE, __HIP_MEMORY_SCOPE_AGENT) == e) {
        __builtin_amdgcn_s_sleep(2);
      }
    }
  }
  __syncthreads();
}

// ---------------------------------------------------------------------------
// Kernel 3: sequential LSTM scan. 64 persistent blocks x 256 threads.
// Block w owns hidden indices u in [w*16, w*16+16): all 4 gate rows local.
// Each row's 1024-dot split across 4 adjacent lanes (256 K each), reduced
// with wave32 shfl_xor. h staged in LDS (broadcast reads), double-buffered
// in global, bf16 W_hh streamed from L2 every step.
// ---------------------------------------------------------------------------
__global__ void lstm_scan(const float* __restrict__ xg,
                          const unsigned short* __restrict__ whh,
                          const int* __restrict__ tok,
                          float* __restrict__ hbuf,
                          float* __restrict__ out,
                          unsigned* __restrict__ bar) {
  __shared__ float sh_h[HIDDEN];
  __shared__ float sh_g[4][16];

  const int tid    = threadIdx.x;
  const int w      = blockIdx.x;       // 64 blocks
  const int sub    = tid & 3;          // K slice (256 elems)
  const int ulocal = (tid >> 2) & 15;  // hidden index within block
  const int gate   = tid >> 6;         // 0..3 -> i,f,g,o
  const int u      = w * 16 + ulocal;
  const int row    = gate * HIDDEN + u;
  const uint4* wrow = (const uint4*)(whh + (size_t)row * HIDDEN) + sub * 32; // 32 uint4 = 256 bf16
  const unsigned nwg = gridDim.x;

  float c_reg = 0.0f, h_reg = 0.0f;    // live state for tid < 16

  for (int t = 0; t < SEQ; ++t) {
    const int cur = t & 1;
    const float* hin = hbuf + cur * HIDDEN;
    for (int idx = tid; idx < HIDDEN; idx += 256) sh_h[idx] = hin[idx];
    __syncthreads();

    const bool m = (tok[t] != 0);
    if (m) {
      float acc = 0.0f;
      const float* hseg = sh_h + sub * 256;
#pragma unroll 4
      for (int kk = 0; kk < 32; ++kk) {
        uint4 q = wrow[kk];
        const float* hp = hseg + kk * 8;
        acc = fmaf(bf_lo(q.x), hp[0], acc);
        acc = fmaf(bf_hi(q.x), hp[1], acc);
        acc = fmaf(bf_lo(q.y), hp[2], acc);
        acc = fmaf(bf_hi(q.y), hp[3], acc);
        acc = fmaf(bf_lo(q.z), hp[4], acc);
        acc = fmaf(bf_hi(q.z), hp[5], acc);
        acc = fmaf(bf_lo(q.w), hp[6], acc);
        acc = fmaf(bf_hi(q.w), hp[7], acc);
      }
      acc += __shfl_xor(acc, 1, 32);
      acc += __shfl_xor(acc, 2, 32);
      if (sub == 0) sh_g[gate][ulocal] = acc + xg[(size_t)t * G4H + row];
    }
    __syncthreads();

    if (tid < 16) {
      if (m) {
        float gi = sh_g[0][tid], gf = sh_g[1][tid], gg = sh_g[2][tid], go = sh_g[3][tid];
        gi = 1.0f / (1.0f + __expf(-gi));
        gf = 1.0f / (1.0f + __expf(-gf));
        go = 1.0f / (1.0f + __expf(-go));
        gg = tanhf(gg);
        c_reg = gf * c_reg + gi * gg;
        h_reg = go * tanhf(c_reg);
      }
      hbuf[(cur ^ 1) * HIDDEN + (w * 16 + tid)] = h_reg;  // publish (unchanged if masked)
    }
    gbar(bar, nwg);
  }

  if (tid < 16) {
    out[w * 16 + tid]          = h_reg;   // h
    out[HIDDEN + w * 16 + tid] = c_reg;   // c
  }
}

// ---------------------------------------------------------------------------
extern "C" void kernel_launch(void* const* d_in, const int* in_sizes, int n_in,
                              void* d_out, int out_size, void* d_ws, size_t ws_size,
                              hipStream_t stream) {
  (void)in_sizes; (void)n_in; (void)out_size; (void)ws_size;
  const int*   tokens = (const int*)  d_in[0];
  const float* emb    = (const float*)d_in[1];
  const float* Wih    = (const float*)d_in[2];
  const float* Whh    = (const float*)d_in[3];
  const float* bih    = (const float*)d_in[4];
  const float* bhh    = (const float*)d_in[5];

  char* ws = (char*)d_ws;
  float*          xg    = (float*)         (ws + OFF_XG);
  unsigned short* xbf   = (unsigned short*)(ws + OFF_X);
  unsigned short* wihbf = (unsigned short*)(ws + OFF_WIH);
  unsigned short* whhbf = (unsigned short*)(ws + OFF_WHH);
  float*          bias  = (float*)         (ws + OFF_BIAS);
  float*          hbuf  = (float*)         (ws + OFF_H);
  unsigned*       bar   = (unsigned*)      (ws + OFF_BAR);

  lstm_prep<<<2048, 256, 0, stream>>>(tokens, emb, Wih, Whh, bih, bhh,
                                      xbf, wihbf, whhbf, bias, hbuf, bar);

  dim3 ggrid(SEQ / 32, G4H / 256);
  lstm_gemm_xg<<<ggrid, 256, 0, stream>>>(xbf, wihbf, bias, xg);

  lstm_scan<<<64, 256, 0, stream>>>(xg, whhbf, tokens, hbuf, (float*)d_out, bar);
}
